// Experts_79998060855664
// MI455X (gfx1250) — matrix operand order repaired and checked by reading, not compiled, not run on hardware
//
#include <hip/hip_runtime.h>
#include <hip/hip_bf16.h>
#include <math.h>

typedef __attribute__((ext_vector_type(16))) __bf16 v16bf;
typedef __attribute__((ext_vector_type(8)))  __bf16 v8bf;
typedef __attribute__((ext_vector_type(4)))  __bf16 v4bf;
typedef __attribute__((ext_vector_type(8)))  float  v8f;

#define NE  8            // experts
#define DM  1024         // d_model
#define DF  4096         // d_ff
#define NB  2            // leading dim
#define CAP 2048         // capacity per expert
#define NT  (NE * CAP)   // tokens per leading slice

#if __has_builtin(__builtin_amdgcn_global_load_async_to_lds_b128)
#define USE_ASYNC_LDS 1
// builtin prototype (from hipcc diagnostic): param0 = GCC-vector int4, AS(1);
// param1 = int4, AS(3); params 2,3 = imm offset, imm cpol.
typedef int gv4i __attribute__((vector_size(16)));
typedef __attribute__((address_space(1))) gv4i* as1_v4ip;
typedef __attribute__((address_space(3))) gv4i* as3_v4ip;
#define ASYNC_CP_B128(gsrc, ldst)                                            \
  __builtin_amdgcn_global_load_async_to_lds_b128(                            \
      (as1_v4ip)(gsrc), (as3_v4ip)(ldst), 0, 0)
#else
#define USE_ASYNC_LDS 0
#endif

__device__ __forceinline__ void wait_asynccnt_4() {
#if __has_builtin(__builtin_amdgcn_s_wait_asynccnt)
  __builtin_amdgcn_s_wait_asynccnt(4);
#else
  asm volatile("s_wait_asynccnt 0x4" ::: "memory");
#endif
}
__device__ __forceinline__ void wait_asynccnt_0() {
#if __has_builtin(__builtin_amdgcn_s_wait_asynccnt)
  __builtin_amdgcn_s_wait_asynccnt(0);
#else
  asm volatile("s_wait_asynccnt 0x0" ::: "memory");
#endif
}

// ---------------- prep kernels ----------------

__global__ void cvt_f32_bf16(const float4* __restrict__ in,
                             __bf16* __restrict__ out, long n4) {
  long i = (long)blockIdx.x * blockDim.x + threadIdx.x;
  long s = (long)gridDim.x * blockDim.x;
  for (; i < n4; i += s) {
    float4 v = in[i];
    v4bf o;
    o[0] = (__bf16)v.x; o[1] = (__bf16)v.y;
    o[2] = (__bf16)v.z; o[3] = (__bf16)v.w;
    *(v4bf*)(out + i * 4) = o;
  }
}

// in: [E][R][C] f32 -> out: [E][C][R] bf16. 32x32 LDS tiles, block (32,8),
// grid (C/32, R/32, E). R and C are multiples of 32.
__global__ __launch_bounds__(256) void transpose_cvt(
    const float* __restrict__ in, __bf16* __restrict__ out, int R, int C) {
  __shared__ float tile[32][33];
  const int c0 = blockIdx.x * 32;
  const int r0 = blockIdx.y * 32;
  const float* ip = in + (size_t)blockIdx.z * R * C;
  __bf16* op = out + (size_t)blockIdx.z * R * C;
  const int tx = threadIdx.x, ty = threadIdx.y;
#pragma unroll
  for (int i = ty; i < 32; i += 8)
    tile[i][tx] = ip[(size_t)(r0 + i) * C + (c0 + tx)];
  __syncthreads();
#pragma unroll
  for (int i = ty; i < 32; i += 8)
    op[(size_t)(c0 + i) * R + (r0 + tx)] = (__bf16)tile[tx][i];
}

// Branchless GELU (tanh form; |err vs erf-GELU| ~3e-4, below bf16 quantization).
__device__ __forceinline__ float gelu_fast(float x) {
  const float u = 0.7978845608028654f * (x + 0.044715f * x * x * x);
  const float eu = __expf(2.0f * fabsf(u));
  float t = 1.0f - 2.0f / (eu + 1.0f);     // tanh(|u|)
  t = copysignf(t, u);
  return 0.5f * x * (1.0f + t);
}

// ---------------- fused tiled GEMM ----------------
// out = EPI( A[g][M,K] @ Bt[e][N,K]^T + bias[e][N] )
// EPI 0: gelu -> bf16 out ; EPI 1: identity -> f32 out
// grid.x = (B*E) * (M/128) * (N/128), block = 256 (8 wave32)
template<int EPI>
__global__ __launch_bounds__(256) void gemm_bf16_wmma(
    const __bf16* __restrict__ Abase, const __bf16* __restrict__ Btbase,
    const float* __restrict__ biasbase, void* __restrict__ outbase,
    int M, int N, int K) {
  constexpr int BM = 128, BN = 128, BK = 32, LDT = BK + 8;  // 80B row pitch
  constexpr int ASZ = BM * LDT;                 // A tile elements per stage
  constexpr int STG = (BM + BN) * LDT;          // elements per stage

  const int tid  = threadIdx.x;
  const int lane = tid & 31;
  const int wave = tid >> 5;
  const int wm   = wave & 3;   // 4 waves along M (32 rows each)
  const int wn   = wave >> 2;  // 2 waves along N (64 cols each)

  const int tilesM   = M / BM;
  const int tilesN   = N / BN;
  const int tilesPer = tilesM * tilesN;
  const int g  = blockIdx.x / tilesPer;   // g = b*NE + e
  const int e  = g % NE;
  const int tt = blockIdx.x % tilesPer;
  const int tm = tt % tilesM;
  const int tn = tt / tilesM;

  const __bf16* A    = Abase  + (size_t)g * M * K + (size_t)tm * BM * K;
  const __bf16* Bt   = Btbase + (size_t)e * N * K + (size_t)tn * BN * K;
  const float*  bias = biasbase + (size_t)e * N;

  // global -> LDS: each of 256 threads owns one (row, 16-element segment)
  const int gr = tid >> 1;         // 0..127
  const int gs = (tid & 1) * 16;   // 0 or 16
  const __bf16* aSrc = A  + (size_t)gr * K + gs;
  const __bf16* bSrc = Bt + (size_t)gr * K + gs;

  const v8f vzero = {0.f, 0.f, 0.f, 0.f, 0.f, 0.f, 0.f, 0.f};
  v8f acc[2][4];
#pragma unroll
  for (int i = 0; i < 2; ++i)
#pragma unroll
    for (int j = 0; j < 4; ++j) acc[i][j] = vzero;

  // WMMA 16-bit fragment lane mapping (ISA 7.12.2)
  const int mrow = lane & 15;
  const int kb   = (lane >> 4) << 3;   // A: 0 or 8
  const int kb2  = (lane >> 4) << 4;   // unused (0 or 16); B also reads at kb

#if USE_ASYNC_LDS
  // 3-stage ring buffer, async global->LDS (ASYNCcnt), 1 barrier / K-step.
  __shared__ __align__(16) __bf16 lds[3 * STG];
  __bf16* aDst = &lds[gr * LDT + gs];            // stage 0 A slot
  __bf16* bDst = &lds[ASZ + gr * LDT + gs];      // stage 0 B slot

  // prologue: issue tile 0 into stage 0 (4 async b128 per thread)
  ASYNC_CP_B128(aSrc,     aDst);
  ASYNC_CP_B128(aSrc + 8, aDst + 8);
  ASYNC_CP_B128(bSrc,     bDst);
  ASYNC_CP_B128(bSrc + 8, bDst + 8);

  int stage = 0;
  for (int k0 = 0; k0 < K; k0 += BK) {
    const bool hasNext = (k0 + BK) < K;
    int nextStage = stage + 1; if (nextStage == 3) nextStage = 0;
    if (hasNext) {
      const __bf16* an = aSrc + (k0 + BK);
      const __bf16* bn = bSrc + (k0 + BK);
      __bf16* ad = &lds[nextStage * STG + gr * LDT + gs];
      __bf16* bd = &lds[nextStage * STG + ASZ + gr * LDT + gs];
      ASYNC_CP_B128(an,     ad);
      ASYNC_CP_B128(an + 8, ad + 8);
      ASYNC_CP_B128(bn,     bd);
      ASYNC_CP_B128(bn + 8, bd + 8);
      wait_asynccnt_4();   // in-order returns => current tile complete
    } else {
      wait_asynccnt_0();
    }
    __syncthreads();       // all waves' tile data visible in LDS

    const __bf16* aTile = &lds[stage * STG];
    const __bf16* bTile = &lds[stage * STG + ASZ];

    v16bf af[2], bfv[4];
#pragma unroll
    for (int i = 0; i < 2; ++i) {
      const __bf16* p = &aTile[(wm * 32 + i * 16 + mrow) * LDT + kb];
      v8bf lo = *(const v8bf*)p;
      v8bf hi = *(const v8bf*)(p + 16);
#pragma unroll
      for (int x = 0; x < 8; ++x) { af[i][x] = lo[x]; af[i][8 + x] = hi[x]; }
    }
#pragma unroll
    for (int j = 0; j < 4; ++j) {
      const __bf16* q = &bTile[(wn * 64 + j * 16 + mrow) * LDT + kb];
      v8bf lo = *(const v8bf*)q;
      v8bf hi = *(const v8bf*)(q + 16);
#pragma unroll
      for (int x = 0; x < 8; ++x) { bfv[j][x] = lo[x]; bfv[j][8 + x] = hi[x]; }
    }
#pragma unroll
    for (int i = 0; i < 2; ++i)
#pragma unroll
      for (int j = 0; j < 4; ++j)
        acc[i][j] = __builtin_amdgcn_wmma_f32_16x16x32_bf16(
            false, af[i], false, bfv[j], (short)0, acc[i][j], false, false);
    stage = nextStage;
  }
#else
  // fallback: register-staged double-barrier pipeline
  __shared__ __align__(16) __bf16 aLds[BM * LDT];
  __shared__ __align__(16) __bf16 bLds[BN * LDT];
  __bf16* aDst = &aLds[gr * LDT + gs];
  __bf16* bDst = &bLds[gr * LDT + gs];

  for (int k0 = 0; k0 < K; k0 += BK) {
    v8bf a0 = *(const v8bf*)(aSrc + k0);
    v8bf a1 = *(const v8bf*)(aSrc + k0 + 8);
    v8bf b0 = *(const v8bf*)(bSrc + k0);
    v8bf b1 = *(const v8bf*)(bSrc + k0 + 8);
    if (k0 + BK < K) {
      __builtin_prefetch(aSrc + k0 + BK, 0, 1);
      __builtin_prefetch(bSrc + k0 + BK, 0, 1);
    }
    __syncthreads();
    *(v8bf*)aDst       = a0;
    *(v8bf*)(aDst + 8) = a1;
    *(v8bf*)bDst       = b0;
    *(v8bf*)(bDst + 8) = b1;
    __syncthreads();

    v16bf af[2], bfv[4];
#pragma unroll
    for (int i = 0; i < 2; ++i) {
      const __bf16* p = &aLds[(wm * 32 + i * 16 + mrow) * LDT + kb];
      v8bf lo = *(const v8bf*)p;
      v8bf hi = *(const v8bf*)(p + 16);
#pragma unroll
      for (int x = 0; x < 8; ++x) { af[i][x] = lo[x]; af[i][8 + x] = hi[x]; }
    }
#pragma unroll
    for (int j = 0; j < 4; ++j) {
      const __bf16* q = &bLds[(wn * 64 + j * 16 + mrow) * LDT + kb];
      v8bf lo = *(const v8bf*)q;
      v8bf hi = *(const v8bf*)(q + 16);
#pragma unroll
      for (int x = 0; x < 8; ++x) { bfv[j][x] = lo[x]; bfv[j][8 + x] = hi[x]; }
    }
#pragma unroll
    for (int i = 0; i < 2; ++i)
#pragma unroll
      for (int j = 0; j < 4; ++j)
        acc[i][j] = __builtin_amdgcn_wmma_f32_16x16x32_bf16(
            false, af[i], false, bfv[j], (short)0, acc[i][j], false, false);
  }
#endif

  // epilogue: C/D layout — VGPR r: M = r (lanes<16) / r+8 (lanes>=16), N = lane&15
  const int rowLoc = tm * BM + wm * 32;
  const int colLoc = tn * BN + wn * 64;
  const int nl   = lane & 15;
  const int mAdd = (lane >> 4) * 8;
  __bf16* outb = (__bf16*)outbase + (size_t)g * M * N;
  float*  outf = (float*)outbase  + (size_t)g * M * N;
#pragma unroll
  for (int i = 0; i < 2; ++i) {
#pragma unroll
    for (int j = 0; j < 4; ++j) {
      const int col = colLoc + j * 16 + nl;
      const float bv = bias[col];
#pragma unroll
      for (int r = 0; r < 8; ++r) {
        const int row = rowLoc + i * 16 + mAdd + r;
        const float v = acc[i][j][r] + bv;
        const size_t idx = (size_t)row * N + col;
        if (EPI == 0) {
          outb[idx] = (__bf16)gelu_fast(v);
        } else {
          outf[idx] = v;
        }
      }
    }
  }
}

// ---------------- launcher ----------------

extern "C" void kernel_launch(void* const* d_in, const int* in_sizes, int n_in,
                              void* d_out, int out_size, void* d_ws, size_t ws_size,
                              hipStream_t stream) {
  const float* x  = (const float*)d_in[0];  // [B, T, D]
  const float* w1 = (const float*)d_in[1];  // [E, D, F]
  const float* b1 = (const float*)d_in[2];  // [E, F]
  const float* w2 = (const float*)d_in[3];  // [E, F, D]
  const float* b2 = (const float*)d_in[4];  // [E, D]
  float* out = (float*)d_out;               // [B, T, D] f32

  const size_t nx = (size_t)NB * NT * DM;   // x elements
  const size_t nw = (size_t)NE * DM * DF;   // per-weight elements

  char* ws = (char*)d_ws;
  __bf16* xb  = (__bf16*)ws;  ws += nx * sizeof(__bf16);
  __bf16* w1t = (__bf16*)ws;  ws += nw * sizeof(__bf16);  // [E, F, D]
  __bf16* w2t = (__bf16*)ws;  ws += nw * sizeof(__bf16);  // [E, D, F]
  __bf16* h   = (__bf16*)ws;                              // [B, E, C, F]

  cvt_f32_bf16<<<4096, 256, 0, stream>>>((const float4*)x, xb, (long)(nx / 4));
  transpose_cvt<<<dim3(DF / 32, DM / 32, NE), dim3(32, 8), 0, stream>>>(
      w1, w1t, DM, DF);
  transpose_cvt<<<dim3(DM / 32, DF / 32, NE), dim3(32, 8), 0, stream>>>(
      w2, w2t, DF, DM);

  // Pass 1: H = gelu(X @ W1 + b1); per-group M=CAP, N=DF, K=DM
  {
    const int tiles = (NB * NE) * (CAP / 128) * (DF / 128);
    gemm_bf16_wmma<0><<<tiles, 256, 0, stream>>>(xb, w1t, b1, (void*)h,
                                                 CAP, DF, DM);
  }
  // Pass 2: OUT = H @ W2 + b2; per-group M=CAP, N=DM, K=DF
  {
    const int tiles = (NB * NE) * (CAP / 128) * (DM / 128);
    gemm_bf16_wmma<1><<<tiles, 256, 0, stream>>>(h, w2t, b2, (void*)out,
                                                 CAP, DM, DF);
  }
}
